// GCNModelWithEdgeAFPreadout_66846870995610
// MI455X (gfx1250) — matrix operand order repaired and checked by reading, not compiled, not run on hardware
//
#include <hip/hip_runtime.h>

#define NPG  32
#define NG   2048
#define NV   65536
#define NE   262144
#define HH   200
#define NL   5
#define TT   2
#define H3   600

typedef __attribute__((ext_vector_type(16))) __bf16 v16bf;
typedef __attribute__((ext_vector_type(8)))  float  v8f;

enum { EPI_PLAIN = 0, EPI_RELU_RES = 1, EPI_ELU = 2 };

// ---------------------------------------------------------------------------
// Wave reductions (wave32)
// ---------------------------------------------------------------------------
__device__ __forceinline__ float wave_sum(float v) {
#pragma unroll
  for (int o = 16; o > 0; o >>= 1) v += __shfl_xor(v, o, 32);
  return v;
}
__device__ __forceinline__ float wave_max(float v) {
#pragma unroll
  for (int o = 16; o > 0; o >>= 1) v = fmaxf(v, __shfl_xor(v, o, 32));
  return v;
}
__device__ __forceinline__ float sigmoidf(float x) { return 1.f / (1.f + __expf(-x)); }

// ---------------------------------------------------------------------------
// Pack a KxN fp32 matrix (optionally transposed source) into bf16 WMMA B
// fragments, fragment order: [ktile][ntile][lane 0..31][elem 0..15].
// B 32x16 bf16 layout: lanes 0-15 hold K = kt*32 + e (e0..7) / kt*32 + 16 + (e-8) (e8..15)
// for col n=lane; lanes 16-31 same with +8 on both K halves, col n=lane-16.
// ---------------------------------------------------------------------------
__global__ __launch_bounds__(256) void pack_b_kernel(
    const float* __restrict__ W, int Ksrc, int Nsrc, int ldsrc, int transpose,
    __bf16* __restrict__ out, int ktiles, int ntiles) {
  int idx = blockIdx.x * blockDim.x + threadIdx.x;
  int total = ktiles * ntiles * 512;
  if (idx >= total) return;
  int e    = idx & 15;
  int l    = (idx >> 4) & 31;
  int tile = idx >> 9;
  int nt   = tile % ntiles;
  int kt   = tile / ntiles;
  int k = kt * 32 + (e & 7) + ((l < 16) ? 0 : 8) + ((e & 8) ? 16 : 0);
  int n = nt * 16 + (l & 15);
  float v = 0.f;
  if (k < Ksrc && n < Nsrc)
    v = transpose ? W[(size_t)n * ldsrc + k] : W[(size_t)k * ldsrc + n];
  out[idx] = (__bf16)v;
}

// ---------------------------------------------------------------------------
// Generic WMMA GEMM: out[M,Nreal] = epilogue(A[M,Ka] * B + bias)
// A fp32 row-major (converted to bf16 fragments on the fly), B pre-packed bf16.
// One wave per 16-row strip; grid*8 waves == M/16 exactly (no row guard).
// A 16x32 bf16 fragment layout (ISA 7.12.2): lane<16 row=lane, e0..7 -> K+0..7,
// e8..15 -> K+16..23 ; lane>=16 row=lane-16, +8 on both K halves.
// ---------------------------------------------------------------------------
template <int EPI, bool ALIGN4, int KT>
__global__ __launch_bounds__(256) void wmma_gemm_kernel(
    const float* __restrict__ A, int lda, int Ka,
    const __bf16* __restrict__ Bpack, int ntiles,
    const float* __restrict__ bias,
    float* __restrict__ out, int ldout, int Nreal,
    const float* __restrict__ resid, int ldres) {
  int wave = blockIdx.x * (blockDim.x >> 5) + (threadIdx.x >> 5);
  int lane = threadIdx.x & 31;
  int r15  = lane & 15;
  int row  = wave * 16 + r15;
  int koff = (lane < 16) ? 0 : 8;
  const float* pr = A + (size_t)row * lda;

  v16bf af[KT];
#pragma unroll
  for (int kt = 0; kt < KT; ++kt) {
#pragma unroll
    for (int c = 0; c < 2; ++c) {
      int ks = kt * 32 + koff + c * 16;
      if (ALIGN4 && (ks + 8 <= Ka)) {
        const float4 q0 = *reinterpret_cast<const float4*>(pr + ks);
        const float4 q1 = *reinterpret_cast<const float4*>(pr + ks + 4);
        af[kt][c * 8 + 0] = (__bf16)q0.x; af[kt][c * 8 + 1] = (__bf16)q0.y;
        af[kt][c * 8 + 2] = (__bf16)q0.z; af[kt][c * 8 + 3] = (__bf16)q0.w;
        af[kt][c * 8 + 4] = (__bf16)q1.x; af[kt][c * 8 + 5] = (__bf16)q1.y;
        af[kt][c * 8 + 6] = (__bf16)q1.z; af[kt][c * 8 + 7] = (__bf16)q1.w;
      } else {
#pragma unroll
        for (int j = 0; j < 8; ++j) {
          float v = (ks + j < Ka) ? pr[ks + j] : 0.f;
          af[kt][c * 8 + j] = (__bf16)v;
        }
      }
    }
  }

  const v16bf* Bp = reinterpret_cast<const v16bf*>(Bpack);
  int mrow0 = wave * 16 + ((lane < 16) ? 0 : 8);
  for (int nt = 0; nt < ntiles; ++nt) {
    // Load all B fragments for this N-tile first: one clause + one wait,
    // then an uninterrupted WMMA chain.
    v16bf bf[KT];
#pragma unroll
    for (int kt = 0; kt < KT; ++kt)
      bf[kt] = Bp[((size_t)kt * ntiles + nt) * 32 + lane];
    v8f acc = {};
#pragma unroll
    for (int kt = 0; kt < KT; ++kt)
      acc = __builtin_amdgcn_wmma_f32_16x16x32_bf16(
          false, af[kt], false, bf[kt], (short)0, acc, false, false);

    int n = nt * 16 + r15;
    if (n < Nreal) {
      float b = bias[n];
      float* po = out + (size_t)mrow0 * ldout + n;
      if (EPI == EPI_RELU_RES) {
        const float* pq = resid + (size_t)mrow0 * ldres + n;
#pragma unroll
        for (int r = 0; r < 8; ++r) {
          *po = fmaxf(acc[r] + b, 0.f) + *pq;
          po += ldout; pq += ldres;
        }
      } else if (EPI == EPI_ELU) {
#pragma unroll
        for (int r = 0; r < 8; ++r) {
          float v = acc[r] + b;
          *po = (v > 0.f) ? v : (__expf(v) - 1.f);
          po += ldout;
        }
      } else {
#pragma unroll
        for (int r = 0; r < 8; ++r) {
          *po = acc[r] + b;
          po += ldout;
        }
      }
    }
  }
}

// ---------------------------------------------------------------------------
// Edge-softmax aggregation. Edges of node v are 4v..4v+3 (dst-sorted, DEG=4).
// agg[v,f] = sum_j a_j * m_j  with m_j = h[src[4v+j],f]+eh[4v+j,f],
// a = softmax_j(m_j).  One block per node, thread per feature.
// ---------------------------------------------------------------------------
__global__ __launch_bounds__(256) void agg_kernel(
    const float* __restrict__ h, const float* __restrict__ eh,
    const int* __restrict__ src, float* __restrict__ agg) {
  int v = blockIdx.x;
  int f = threadIdx.x;
  if (f >= HH) return;
  int e0 = v * 4;
  int s0 = src[e0], s1 = src[e0 + 1], s2 = src[e0 + 2], s3 = src[e0 + 3];
  float m0 = h[(size_t)s0 * HH + f] + eh[(size_t)(e0 + 0) * HH + f];
  float m1 = h[(size_t)s1 * HH + f] + eh[(size_t)(e0 + 1) * HH + f];
  float m2 = h[(size_t)s2 * HH + f] + eh[(size_t)(e0 + 2) * HH + f];
  float m3 = h[(size_t)s3 * HH + f] + eh[(size_t)(e0 + 3) * HH + f];
  float mx = fmaxf(fmaxf(m0, m1), fmaxf(m2, m3));
  float x0 = __expf(m0 - mx), x1 = __expf(m1 - mx);
  float x2 = __expf(m2 - mx), x3 = __expf(m3 - mx);
  float den = x0 + x1 + x2 + x3;
  agg[(size_t)v * HH + f] = (m0 * x0 + m1 * x1 + m2 * x2 + m3 * x3) / den;
}

// gf[g,f] = sum over 32 contiguous nodes
__global__ __launch_bounds__(256) void graph_sum_kernel(
    const float* __restrict__ h, float* __restrict__ gf) {
  int g = blockIdx.x;
  int f = threadIdx.x;
  if (f >= HH) return;
  float s = 0.f;
#pragma unroll 8
  for (int j = 0; j < NPG; ++j) s += h[((size_t)g * NPG + j) * HH + f];
  gf[(size_t)g * HH + f] = s;
}

// per-graph scalar: sg[g] = sum_f relu(gf[g,f]) * lgW[f]   (wave per graph)
__global__ __launch_bounds__(256) void sg_kernel(
    const float* __restrict__ gf, const float* __restrict__ lgW,
    float* __restrict__ sg) {
  int g = (blockIdx.x * blockDim.x + threadIdx.x) >> 5;
  int lane = threadIdx.x & 31;
  if (g >= NG) return;
  float s = 0.f;
  for (int f = lane; f < HH; f += 32) s += fmaxf(gf[(size_t)g * HH + f], 0.f) * lgW[f];
  s = wave_sum(s);
  if (lane == 0) sg[g] = s;
}

// z[v] = leaky_relu( sg[graph] + sum_f h[v,f]*lgW[HH+f] + lgb )  (wave per node)
__global__ __launch_bounds__(256) void z_kernel(
    const float* __restrict__ h, const float* __restrict__ sg,
    const float* __restrict__ lgW, const float* __restrict__ lgb,
    float* __restrict__ z) {
  int v = (blockIdx.x * blockDim.x + threadIdx.x) >> 5;
  int lane = threadIdx.x & 31;
  if (v >= NV) return;
  float s = 0.f;
  for (int f = lane; f < HH; f += 32) s += h[(size_t)v * HH + f] * lgW[HH + f];
  s = wave_sum(s);
  if (lane == 0) {
    float x = s + sg[v / NPG] + lgb[0];
    z[v] = (x > 0.f) ? x : 0.01f * x;
  }
}

// per-graph softmax over 32 nodes: lane == node-in-graph (NPG == wave32)
__global__ __launch_bounds__(256) void attn_kernel(
    const float* __restrict__ z, float* __restrict__ a) {
  int g = (blockIdx.x * blockDim.x + threadIdx.x) >> 5;
  int lane = threadIdx.x & 31;
  if (g >= NG) return;
  int v = g * NPG + lane;
  float zv = z[v];
  float mx = wave_max(zv);
  float e = __expf(zv - mx);
  float den = wave_sum(e);
  a[v] = e / den;
}

// hbar[g,f] = sum_j a[g*32+j] * h[g*32+j, f]
__global__ __launch_bounds__(256) void hbar_kernel(
    const float* __restrict__ h, const float* __restrict__ a,
    float* __restrict__ hbar) {
  int g = blockIdx.x;
  int f = threadIdx.x;
  if (f >= HH) return;
  float s = 0.f;
#pragma unroll 8
  for (int j = 0; j < NPG; ++j)
    s += a[g * NPG + j] * h[((size_t)g * NPG + j) * HH + f];
  hbar[(size_t)g * HH + f] = s;
}

// GRU gates (torch order r,z,n): gf' = (1-u)*n + u*gf
__global__ __launch_bounds__(256) void gru_kernel(
    const float* __restrict__ gi, const float* __restrict__ gh,
    const float* __restrict__ gfin, float* __restrict__ gfout) {
  int idx = blockIdx.x * blockDim.x + threadIdx.x;
  if (idx >= NG * HH) return;
  int g = idx / HH, f = idx % HH;
  const float* pi = gi + (size_t)g * H3;
  const float* ph = gh + (size_t)g * H3;
  float r = sigmoidf(pi[f] + ph[f]);
  float u = sigmoidf(pi[HH + f] + ph[HH + f]);
  float n = tanhf(pi[2 * HH + f] + r * ph[2 * HH + f]);
  gfout[idx] = (1.f - u) * n + u * gfin[idx];
}

// ---------------------------------------------------------------------------
extern "C" void kernel_launch(void* const* d_in, const int* in_sizes, int n_in,
                              void* d_out, int out_size, void* d_ws, size_t ws_size,
                              hipStream_t stream) {
  (void)in_sizes; (void)n_in; (void)out_size; (void)ws_size;
  const float* node_feat = (const float*)d_in[0];
  const float* edge_feat = (const float*)d_in[1];
  const int*   src       = (const int*)d_in[2];
  // d_in[3]=dst, d_in[4]=graph_id : structure is fixed (dst-sorted, contiguous graphs)
  const float* node_W = (const float*)d_in[5];
  const float* node_b = (const float*)d_in[6];
  const float* edge_W = (const float*)d_in[7];
  const float* edge_b = (const float*)d_in[8];
  const float* gnn_W  = (const float*)d_in[9];
  const float* gnn_b  = (const float*)d_in[10];
  const float* lg_W   = (const float*)d_in[11];
  const float* lg_b   = (const float*)d_in[12];
  const float* pr_W   = (const float*)d_in[13];
  const float* pr_b   = (const float*)d_in[14];
  const float* W_ih   = (const float*)d_in[15];
  const float* W_hh   = (const float*)d_in[16];
  const float* b_ih   = (const float*)d_in[17];
  const float* b_hh   = (const float*)d_in[18];

  // ---- workspace carve-out (256B aligned regions) ----
  char* base = (char*)d_ws;
  size_t off = 0;
  auto alloc = [&](size_t bytes) -> void* {
    off = (off + 255) & ~(size_t)255;
    void* p = base + off;
    off += bytes;
    return p;
  };
  float* h    = (float*)alloc((size_t)NV * HH * 4);
  float* eh   = (float*)alloc((size_t)NE * HH * 4);
  float* agg  = (float*)alloc((size_t)NV * HH * 4);
  float* gf   = (float*)alloc((size_t)NG * HH * 4);
  float* zb   = (float*)alloc((size_t)NV * 4);
  float* ab   = (float*)alloc((size_t)NV * 4);
  float* sg   = (float*)alloc((size_t)NG * 4);
  float* hbar = (float*)alloc((size_t)NG * HH * 4);
  float* ctx  = (float*)alloc((size_t)NG * HH * 4);
  float* gi   = (float*)alloc((size_t)NG * H3 * 4);
  float* gh   = (float*)alloc((size_t)NG * H3 * 4);
  __bf16* pNodeW = (__bf16*)alloc((size_t)3 * 13 * 512 * 2);
  __bf16* pEdgeW = (__bf16*)alloc((size_t)1 * 13 * 512 * 2);
  __bf16* pGnn   = (__bf16*)alloc((size_t)NL * 7 * 13 * 512 * 2);
  __bf16* pPr    = (__bf16*)alloc((size_t)TT * 7 * 13 * 512 * 2);
  __bf16* pWih   = (__bf16*)alloc((size_t)TT * 7 * 38 * 512 * 2);
  __bf16* pWhh   = (__bf16*)alloc((size_t)TT * 7 * 38 * 512 * 2);

  auto pack = [&](const float* W, int Ks, int Ns, int lds, int tr,
                  __bf16* out, int kt, int nt) {
    int total = kt * nt * 512;
    pack_b_kernel<<<(total + 255) / 256, 256, 0, stream>>>(W, Ks, Ns, lds, tr, out, kt, nt);
  };
  pack(node_W, 74, HH, HH, 0, pNodeW, 3, 13);
  pack(edge_W, 12, HH, HH, 0, pEdgeW, 1, 13);
  for (int i = 0; i < NL; ++i)
    pack(gnn_W + (size_t)i * HH * HH, HH, HH, HH, 0, pGnn + (size_t)i * 7 * 13 * 512, 7, 13);
  for (int t = 0; t < TT; ++t) {
    pack(pr_W + (size_t)t * HH * HH, HH, HH, HH, 0, pPr + (size_t)t * 7 * 13 * 512, 7, 13);
    pack(W_ih + (size_t)t * H3 * HH, HH, H3, HH, 1, pWih + (size_t)t * 7 * 38 * 512, 7, 38);
    pack(W_hh + (size_t)t * H3 * HH, HH, H3, HH, 1, pWhh + (size_t)t * 7 * 38 * 512, 7, 38);
  }

  // ---- embeddings (WMMA) ----
  // h = node_feat @ node_W + node_b   (K=74, rows unaligned for float4)
  wmma_gemm_kernel<EPI_PLAIN, false, 3><<<NV / 16 / 8, 256, 0, stream>>>(
      node_feat, 74, 74, pNodeW, 13, node_b, h, HH, HH, nullptr, 0);
  // eh = edge_feat @ edge_W + edge_b  (K=12)
  wmma_gemm_kernel<EPI_PLAIN, true, 1><<<NE / 16 / 8, 256, 0, stream>>>(
      edge_feat, 12, 12, pEdgeW, 13, edge_b, eh, HH, HH, nullptr, 0);

  // ---- 5 GCN layers: agg -> h = relu(agg@W + b) + h ----
  for (int i = 0; i < NL; ++i) {
    agg_kernel<<<NV, 256, 0, stream>>>(h, eh, src, agg);
    wmma_gemm_kernel<EPI_RELU_RES, true, 7><<<NV / 16 / 8, 256, 0, stream>>>(
        agg, HH, HH, pGnn + (size_t)i * 7 * 13 * 512, 13,
        gnn_b + (size_t)i * HH, h, HH, HH, h, HH);
  }

  // ---- AttentiveFP readout ----
  graph_sum_kernel<<<NG, 256, 0, stream>>>(h, gf);
  for (int t = 0; t < TT; ++t) {
    const float* lgWt = lg_W + (size_t)t * 2 * HH;
    sg_kernel<<<NG * 32 / 256, 256, 0, stream>>>(gf, lgWt, sg);
    z_kernel<<<NV * 32 / 256, 256, 0, stream>>>(h, sg, lgWt, lg_b + t, zb);
    attn_kernel<<<NG * 32 / 256, 256, 0, stream>>>(zb, ab);
    hbar_kernel<<<NG, 256, 0, stream>>>(h, ab, hbar);
    // ctx = elu(hbar @ pr_W + pr_b)  (exact via linearity: sum a = 1)
    wmma_gemm_kernel<EPI_ELU, true, 7><<<NG / 16 / 8, 256, 0, stream>>>(
        hbar, HH, HH, pPr + (size_t)t * 7 * 13 * 512, 13,
        pr_b + (size_t)t * HH, ctx, HH, HH, nullptr, 0);
    // gi = ctx @ W_ih^T + b_ih ; gh = gf @ W_hh^T + b_hh
    wmma_gemm_kernel<EPI_PLAIN, true, 7><<<NG / 16 / 8, 256, 0, stream>>>(
        ctx, HH, HH, pWih + (size_t)t * 7 * 38 * 512, 38,
        b_ih + (size_t)t * H3, gi, H3, H3, nullptr, 0);
    wmma_gemm_kernel<EPI_PLAIN, true, 7><<<NG / 16 / 8, 256, 0, stream>>>(
        gf, HH, HH, pWhh + (size_t)t * 7 * 38 * 512, 38,
        b_hh + (size_t)t * H3, gh, H3, H3, nullptr, 0);
    gru_kernel<<<(NG * HH + 255) / 256, 256, 0, stream>>>(
        gi, gh, gf, (t == TT - 1) ? (float*)d_out : gf);
  }
}
